// MultiHeadAttention_19516331393694
// MI455X (gfx1250) — matrix operand order repaired and checked by reading, not compiled, not run on hardware
//
#include <hip/hip_runtime.h>

// Problem constants (match reference).
constexpr int Bb  = 4;
constexpr int Ss  = 2048;
constexpr int Dd  = 1024;
constexpr int Hh  = 16;
constexpr int DKV = 64;           // DK == DV == 64
#define NEGV (-1e10f)

typedef __attribute__((ext_vector_type(16))) _Float16 v16h;
typedef __attribute__((ext_vector_type(8)))  float    v8f;

static __device__ __forceinline__ v8f wmma16(v16h a, v16h b, v8f c) {
  // v_wmma_f32_16x16x32_f16  (D = A(16x32) x B(32x16) + C)
  return __builtin_amdgcn_wmma_f32_16x16x32_f16(false, a, false, b, (short)0, c,
                                                false, false);
}

// ---- Fragment loaders (ISA 7.12.2 layouts, wave32) -------------------------
// A (16x32 f16): lane L holds row (L&15); K runs {0..7,16..23} (lanes 0-15)
// and {8..15,24..31} (lanes 16-31). Two 16B loads per lane.
static __device__ __forceinline__ v16h a_frag_f16(const _Float16* base, int stride) {
  unsigned L = threadIdx.x & 31u;
  const _Float16* row = base + (size_t)(L & 15u) * stride + ((L < 16u) ? 0u : 8u);
  union { v16h v; uint4 q[2]; } u;
  u.q[0] = *reinterpret_cast<const uint4*>(row);
  u.q[1] = *reinterpret_cast<const uint4*>(row + 16);
  return u.v;
}

// Same A layout but sourced from row-major f32 (converted to f16 on the fly).
static __device__ __forceinline__ v16h a_frag_f32(const float* base, int stride) {
  unsigned L = threadIdx.x & 31u;
  const float* row = base + (size_t)(L & 15u) * stride + ((L < 16u) ? 0u : 8u);
  union { v16h v; _Float16 h[16]; } u;
#pragma unroll
  for (int i = 0; i < 8; ++i) {
    u.h[i]     = (_Float16)row[i];
    u.h[8 + i] = (_Float16)row[16 + i];
  }
  return u.v;
}

// B (32x16 f16) from storage where the storage ROW index is the matrix column
// N and the storage COLUMN index is K (i.e. B^T row-major). Lane L holds
// column (L&15); K runs {0..7,16..23} (lanes 0-15) and {8..15,24..31} (lanes 16-31). Two 16B loads.
static __device__ __forceinline__ v16h b_frag_f16t(const _Float16* base, int stride) {
  unsigned L = threadIdx.x & 31u;
  const _Float16* row = base + (size_t)(L & 15u) * stride + ((L < 16u) ? 0u : 8u);
  union { v16h v; uint4 q[2]; } u;
  u.q[0] = *reinterpret_cast<const uint4*>(row);
  u.q[1] = *reinterpret_cast<const uint4*>(row + 16);
  return u.v;
}

// ---- CDNA5 async global->LDS staging (ASYNCcnt, ISA 10.x / 15.18.3) --------
// Copy a contiguous 4KB region: 8 x (32 lanes x 16B) async b128 transfers.
static __device__ __forceinline__ void async_copy_contig_4k(unsigned lds_off,
                                                            const _Float16* g) {
  const unsigned L = threadIdx.x & 31u;
  const unsigned l0 = lds_off + L * 16u;
  const unsigned long long g0 = (unsigned long long)(size_t)((const char*)g + L * 16u);
#pragma unroll
  for (int j = 0; j < 8; ++j) {
    unsigned l = l0 + (unsigned)(j * 512);
    unsigned long long gg = g0 + (unsigned long long)(j * 512);
    asm volatile("global_load_async_to_lds_b128 %0, %1, off"
                 :: "v"(l), "v"(gg) : "memory");
  }
}

// Copy 64 row-segments of 64B (V^T tile: 64 rows x 32 f16, row stride S*2 B)
// into LDS laid out [64][32] f16. 4 lanes cover one 64B segment.
static __device__ __forceinline__ void async_copy_vt_tile(unsigned lds_off,
                                                          const _Float16* vt_t0) {
  const unsigned L = threadIdx.x & 31u;
  const unsigned seg = L >> 2;          // 8 rows per instruction
  const unsigned byt = (L & 3u) * 16u;
#pragma unroll
  for (int j = 0; j < 8; ++j) {
    const unsigned row = (unsigned)(j * 8) + seg;          // v-row 0..63
    unsigned l = lds_off + row * 64u + byt;
    unsigned long long g = (unsigned long long)(size_t)
        ((const char*)vt_t0 + (size_t)row * (Ss * 2) + byt);
    asm volatile("global_load_async_to_lds_b128 %0, %1, off"
                 :: "v"(l), "v"(g) : "memory");
  }
}

// ---- Stage 0: weight transpose + f32->f16 ----------------------------------
// Wq/Wk/Wv [H,D,64] -> [H,64,D] f16 ; Wo [1024,D] -> [D,1024] f16.
__global__ void prep_weights_kernel(const float* __restrict__ Wq,
                                    const float* __restrict__ Wk,
                                    const float* __restrict__ Wv,
                                    const float* __restrict__ Wo,
                                    _Float16* __restrict__ wqt,
                                    _Float16* __restrict__ wkt,
                                    _Float16* __restrict__ wvt,
                                    _Float16* __restrict__ wot) {
  const size_t NW = (size_t)Hh * Dd * DKV;                 // 1,048,576
  size_t i = (size_t)blockIdx.x * blockDim.x + threadIdx.x;
  if (i >= NW) return;
  const int h = (int)(i / (Dd * DKV));
  const int r = (int)(i % (Dd * DKV));
  const int d = r / DKV;
  const int n = r % DKV;
  const size_t dst = (size_t)h * DKV * Dd + (size_t)n * Dd + d;
  wqt[dst] = (_Float16)Wq[i];
  wkt[dst] = (_Float16)Wk[i];
  wvt[dst] = (_Float16)Wv[i];
  const int c = (int)(i / Dd), n2 = (int)(i % Dd);         // Wo[c][n2]
  wot[(size_t)n2 * (Hh * DKV) + c] = (_Float16)Wo[i];
}

// ---- Stage 1: fused per-head Q/K/V projection ------------------------------
// grid = (B*S/32, H, 3), block = 32. One wave: 32(s) x 64(k) tile, K = D.
// Q,K stored [B,H,S,64] f16; V stored transposed [B,H,64,S] f16.
__global__ void qkv_proj_kernel(const float* __restrict__ q_in,
                                const float* __restrict__ k_in,
                                const float* __restrict__ v_in,
                                const _Float16* __restrict__ wqt,
                                const _Float16* __restrict__ wkt,
                                const _Float16* __restrict__ wvt,
                                _Float16* __restrict__ qf,
                                _Float16* __restrict__ kf,
                                _Float16* __restrict__ vt) {
  const int st    = blockIdx.x;            // 0 .. B*S/32-1
  const int h     = blockIdx.y;
  const int which = blockIdx.z;            // 0=Q 1=K 2=V
  const int b  = st / (Ss / 32);
  const int s0 = (st % (Ss / 32)) * 32;

  const float* X = (which == 0) ? q_in : (which == 1) ? k_in : v_in;
  const _Float16* W =
      ((which == 0) ? wqt : (which == 1) ? wkt : wvt) + (size_t)h * DKV * Dd;
  const float* Xb = X + ((size_t)b * Ss + s0) * Dd;

  const v8f zero = {};
  v8f acc[2][4];
#pragma unroll
  for (int mt = 0; mt < 2; ++mt)
#pragma unroll
    for (int nt = 0; nt < 4; ++nt) acc[mt][nt] = zero;

  for (int d0 = 0; d0 < Dd; d0 += 32) {
    v16h a0 = a_frag_f32(Xb + d0, Dd);
    v16h a1 = a_frag_f32(Xb + (size_t)16 * Dd + d0, Dd);
#pragma unroll
    for (int nt = 0; nt < 4; ++nt) {
      v16h bw = b_frag_f16t(W + (size_t)(nt * 16) * Dd + d0, Dd);
      acc[0][nt] = wmma16(a0, bw, acc[0][nt]);
      acc[1][nt] = wmma16(a1, bw, acc[1][nt]);
    }
  }

  const unsigned L = threadIdx.x & 31u;
  const unsigned col = L & 15u;
  const unsigned rbase = (L < 16u) ? 0u : 8u;
#pragma unroll
  for (int mt = 0; mt < 2; ++mt) {
    if (which < 2) {
      _Float16* dst = ((which == 0) ? qf : kf) +
                      ((size_t)(b * Hh + h) * Ss + s0 + mt * 16) * DKV;
#pragma unroll
      for (int nt = 0; nt < 4; ++nt)
#pragma unroll
        for (int i = 0; i < 8; ++i)
          dst[(size_t)(rbase + i) * DKV + nt * 16 + col] = (_Float16)acc[mt][nt][i];
    } else {
      _Float16* dst = vt + (size_t)(b * Hh + h) * DKV * Ss;   // [v][s]
#pragma unroll
      for (int nt = 0; nt < 4; ++nt)
#pragma unroll
        for (int i = 0; i < 8; ++i)
          dst[(size_t)(nt * 16 + col) * Ss + s0 + mt * 16 + rbase + i] =
              (_Float16)acc[mt][nt][i];
    }
  }
}

// ---- Stage 2: masked flash attention, online softmax -----------------------
// grid = (S/16, H, B), block = 32. K/V^T tiles double-buffered in LDS via
// async global->LDS copies (ASYNCcnt); P re-laid-out C->A via LDS.
__global__ void attn_kernel(const _Float16* __restrict__ qf,
                            const _Float16* __restrict__ kf,
                            const _Float16* __restrict__ vt,
                            const int* __restrict__ mask,
                            _Float16* __restrict__ attn) {
  __shared__ _Float16 lds_k[2][32 * DKV];   // [tloc][d]   4KB each
  __shared__ _Float16 lds_v[2][DKV * 32];   // [v][tloc]   4KB each
  __shared__ _Float16 lds_p[16 * 32];       // P exchange  1KB

  const int s0 = blockIdx.x * 16;
  const int h  = blockIdx.y;
  const int b  = blockIdx.z;

  const _Float16* Q  = qf + ((size_t)(b * Hh + h) * Ss + s0) * DKV;
  const _Float16* K  = kf + (size_t)(b * Hh + h) * Ss * DKV;
  const _Float16* Vt = vt + (size_t)(b * Hh + h) * DKV * Ss;
  const int*      Mk = mask + (size_t)b * Ss * Ss + (size_t)s0 * Ss;

  const unsigned koff[2] = {(unsigned)(size_t)&lds_k[0][0],
                            (unsigned)(size_t)&lds_k[1][0]};
  const unsigned voff[2] = {(unsigned)(size_t)&lds_v[0][0],
                            (unsigned)(size_t)&lds_v[1][0]};

  const v16h aq0 = a_frag_f16(Q, DKV);        // d = 0..31
  const v16h aq1 = a_frag_f16(Q + 32, DKV);   // d = 32..63

  const unsigned L = threadIdx.x & 31u;
  const unsigned col = L & 15u;
  const unsigned rbase = (L < 16u) ? 0u : 8u;

  const v8f zero = {};
  v8f o[4];
#pragma unroll
  for (int nt = 0; nt < 4; ++nt) o[nt] = zero;
  float m_run[8], l_run[8];
#pragma unroll
  for (int i = 0; i < 8; ++i) { m_run[i] = -1e30f; l_run[i] = 0.0f; }

  // Prologue: stage tile 0.
  async_copy_contig_4k(koff[0], K);
  async_copy_vt_tile(voff[0], Vt);

  int cur = 0;
  for (int t0 = 0; t0 < Ss; t0 += 32) {
    const bool more = (t0 + 32) < Ss;
    if (more) {
      // Kick off tile t0+32 into the other buffer, then wait for tile t0
      // (16 async ops in flight belong to the next tile).
      async_copy_contig_4k(koff[cur ^ 1], K + (size_t)(t0 + 32) * DKV);
      async_copy_vt_tile(voff[cur ^ 1], Vt + t0 + 32);
      __builtin_prefetch(Mk + t0 + 32, 0, 1);
      asm volatile("s_wait_asynccnt 0x10" ::: "memory");
    } else {
      asm volatile("s_wait_asynccnt 0x0" ::: "memory");
    }
    const _Float16* Kt  = lds_k[cur];
    const _Float16* Vtl = lds_v[cur];

    // scores: two 16x16 tiles covering keys t0..t0+31, K-dim = DK = 64
    v8f sa = zero, sb = zero;
    sa = wmma16(aq0, b_frag_f16t(Kt, DKV), sa);
    sa = wmma16(aq1, b_frag_f16t(Kt + 32, DKV), sa);
    sb = wmma16(aq0, b_frag_f16t(Kt + 16 * DKV, DKV), sb);
    sb = wmma16(aq1, b_frag_f16t(Kt + 16 * DKV + 32, DKV), sb);

    // scale + mask + online softmax update (per lane: 8 rows x 2 columns)
#pragma unroll
    for (int i = 0; i < 8; ++i) {
      float x0 = sa[i] * 0.125f;              // 1/sqrt(DK)
      float x1 = sb[i] * 0.125f;
      const size_t mrow = (size_t)(rbase + i) * Ss + t0;
      if (Mk[mrow + col] == 0)      x0 = NEGV;
      if (Mk[mrow + 16 + col] == 0) x1 = NEGV;

      float mx = fmaxf(x0, x1);               // row max across 16 lanes
      mx = fmaxf(mx, __shfl_xor(mx, 1));
      mx = fmaxf(mx, __shfl_xor(mx, 2));
      mx = fmaxf(mx, __shfl_xor(mx, 4));
      mx = fmaxf(mx, __shfl_xor(mx, 8));
      const float mnew = fmaxf(m_run[i], mx);
      const float corr = __expf(m_run[i] - mnew);
      m_run[i] = mnew;

      const float p0 = __expf(x0 - mnew);
      const float p1 = __expf(x1 - mnew);
      float ps = p0 + p1;                     // row sum across 16 lanes
      ps += __shfl_xor(ps, 1);
      ps += __shfl_xor(ps, 2);
      ps += __shfl_xor(ps, 4);
      ps += __shfl_xor(ps, 8);
      l_run[i] = l_run[i] * corr + ps;

#pragma unroll
      for (int nt = 0; nt < 4; ++nt) o[nt][i] *= corr;   // rescale accumulators

      lds_p[(rbase + i) * 32 + col]      = (_Float16)p0; // P -> LDS (C layout)
      lds_p[(rbase + i) * 32 + 16 + col] = (_Float16)p1;
    }
    asm volatile("s_wait_dscnt 0" ::: "memory");          // LDS stores visible

    const v16h ap = a_frag_f16(lds_p, 32);                // P as A (16x32)
#pragma unroll
    for (int nt = 0; nt < 4; ++nt)
      o[nt] = wmma16(ap, b_frag_f16t(Vtl + nt * 16 * 32, 32), o[nt]);

    cur ^= 1;
  }

  // normalize + store head-concat layout [B*S, H*DV] (f16) for stage 3
#pragma unroll
  for (int i = 0; i < 8; ++i) {
    const float inv = 1.0f / l_run[i];
    _Float16* dst = attn + ((size_t)b * Ss + s0 + rbase + i) * (Hh * DKV) + h * DKV;
#pragma unroll
    for (int nt = 0; nt < 4; ++nt)
      dst[nt * 16 + col] = (_Float16)(o[nt][i] * inv);
  }
}

// ---- Stage 3: output projection [B*S,1024](f16) @ Wo -----------------------
// grid = (B*S/32, D/64), block = 32; Wo pre-transposed to f16 [D,1024].
__global__ void out_proj_kernel(const _Float16* __restrict__ attn,
                                const _Float16* __restrict__ wot,
                                float* __restrict__ out) {
  const size_t row0 = (size_t)blockIdx.x * 32;
  const int n0 = blockIdx.y * 64;
  constexpr int C = Hh * DKV;                 // 1024 (reduction dim)

  const v8f zero = {};
  v8f acc[2][4];
#pragma unroll
  for (int mt = 0; mt < 2; ++mt)
#pragma unroll
    for (int nt = 0; nt < 4; ++nt) acc[mt][nt] = zero;

  for (int c0 = 0; c0 < C; c0 += 32) {
    v16h a0 = a_frag_f16(attn + row0 * C + c0, C);
    v16h a1 = a_frag_f16(attn + (row0 + 16) * C + c0, C);
#pragma unroll
    for (int nt = 0; nt < 4; ++nt) {
      v16h bw = b_frag_f16t(wot + (size_t)(n0 + nt * 16) * C + c0, C);
      acc[0][nt] = wmma16(a0, bw, acc[0][nt]);
      acc[1][nt] = wmma16(a1, bw, acc[1][nt]);
    }
  }

  const unsigned L = threadIdx.x & 31u;
  const unsigned col = L & 15u;
  const unsigned rbase = (L < 16u) ? 0u : 8u;
#pragma unroll
  for (int mt = 0; mt < 2; ++mt)
#pragma unroll
    for (int nt = 0; nt < 4; ++nt)
#pragma unroll
      for (int i = 0; i < 8; ++i)
        out[(row0 + mt * 16 + rbase + i) * Dd + n0 + nt * 16 + col] =
            acc[mt][nt][i];
}

extern "C" void kernel_launch(void* const* d_in, const int* in_sizes, int n_in,
                              void* d_out, int out_size, void* d_ws, size_t ws_size,
                              hipStream_t stream) {
  (void)in_sizes; (void)n_in; (void)out_size; (void)ws_size;
  const float* q_in = (const float*)d_in[0];
  const float* k_in = (const float*)d_in[1];
  const float* v_in = (const float*)d_in[2];
  const int*   mask = (const int*)d_in[3];
  const float* Wq   = (const float*)d_in[4];
  const float* Wk   = (const float*)d_in[5];
  const float* Wv   = (const float*)d_in[6];
  const float* Wo   = (const float*)d_in[7];
  float* out = (float*)d_out;

  const size_t perMat = (size_t)Bb * Hh * Ss * DKV;   // 8,388,608 f16 elems
  const size_t perW   = (size_t)Hh * Dd * DKV;        // 1,048,576 f16 elems
  _Float16* qf   = (_Float16*)d_ws;
  _Float16* kf   = qf + perMat;
  _Float16* vt   = kf + perMat;
  _Float16* attn = vt + perMat;
  _Float16* wqt  = attn + perMat;
  _Float16* wkt  = wqt + perW;
  _Float16* wvt  = wkt + perW;
  _Float16* wot  = wvt + perW;                         // ~76 MB workspace

  prep_weights_kernel<<<dim3((unsigned)(perW / 256)), 256, 0, stream>>>(
      Wq, Wk, Wv, Wo, wqt, wkt, wvt, wot);
  qkv_proj_kernel<<<dim3(Bb * Ss / 32, Hh, 3), 32, 0, stream>>>(
      q_in, k_in, v_in, wqt, wkt, wvt, qf, kf, vt);
  attn_kernel<<<dim3(Ss / 16, Hh, Bb), 32, 0, stream>>>(qf, kf, vt, mask, attn);
  out_proj_kernel<<<dim3(Bb * Ss / 32, Dd / 64), 32, 0, stream>>>(attn, wot, out);
}
